// MultiHeadAttention_83099027243130
// MI455X (gfx1250) — matrix operand order repaired and checked by reading, not compiled, not run on hardware
//
#include <hip/hip_runtime.h>

// ---------------------------------------------------------------------------
// Multi-head attention forward for MI455X (gfx1250), bf16 WMMA + flash softmax
// B=2, S=2048, D=1024, H=16, dk=64
// ---------------------------------------------------------------------------

typedef unsigned int u32;
typedef __bf16 bf16;
typedef __attribute__((ext_vector_type(16))) __bf16 bf16x16;
typedef __attribute__((ext_vector_type(8)))  float  f32x8;
typedef __attribute__((ext_vector_type(4)))  u32    u32x4;

#define DEV __device__ __forceinline__

static constexpr int Bn  = 2;
static constexpr int Sn  = 2048;
static constexpr int Dn  = 1024;
static constexpr int Hn  = 16;
static constexpr int DK  = 64;
static constexpr float NEGV  = -1.0e9f;
static constexpr float SCALE = 0.125f;   // 1/sqrt(64)

DEV unsigned short bf16_bits(float f) {
  u32 u = __builtin_bit_cast(u32, f);
  u32 r = u + 0x7fffu + ((u >> 16) & 1u);   // round-to-nearest-even
  return (unsigned short)(r >> 16);
}

DEV bf16 to_bf16(float f) {
  return __builtin_bit_cast(bf16, bf16_bits(f));
}

struct Frag2 { u32x4 a, b; };

DEV bf16x16 pack_frag(u32x4 lo, u32x4 hi) {
  Frag2 t{lo, hi};
  return __builtin_bit_cast(bf16x16, t);
}

// A-matrix fragment (16 rows x K=32), source row-major, ld in elements.
// Lane L: row = L&15; k-chunks {kh*8..+8} and {16+kh*8..+8}, kh = L>>4.
DEV bf16x16 load_a(const bf16* src, int ld) {
  int lane = threadIdx.x & 31;
  const bf16* p = src + (size_t)(lane & 15) * ld + (lane >> 4) * 8;
  u32x4 lo = *reinterpret_cast<const u32x4*>(p);
  u32x4 hi = *reinterpret_cast<const u32x4*>(p + 16);
  return pack_frag(lo, hi);
}

// B-matrix fragment (K=32 x 16 cols), source row-major [N][K] (K contiguous).
// Lane L: col n = L&15; k-chunks {kh*8..+8} and {16+kh*8..+8}, kh = L>>4.
DEV bf16x16 load_b(const bf16* src, int ld) {
  int lane = threadIdx.x & 31;
  const bf16* p = src + (size_t)(lane & 15) * ld + (lane >> 4) * 8;
  u32x4 lo = *reinterpret_cast<const u32x4*>(p);
  u32x4 hi = *reinterpret_cast<const u32x4*>(p + 16);
  return pack_frag(lo, hi);
}

// A-fragment from a per-wave LDS region holding two 16x16 bf16 tiles
// (tile0 = cols 0..15, tile1 = cols 16..31, each row-major stride 16).
DEV bf16x16 load_a_lds(const bf16* base) {
  int lane = threadIdx.x & 31;
  const bf16* p = base + (lane & 15) * 16 + (lane >> 4) * 8;
  u32x4 lo = *reinterpret_cast<const u32x4*>(p);        // tile0 chunk
  u32x4 hi = *reinterpret_cast<const u32x4*>(p + 256);  // tile1 chunk
  return pack_frag(lo, hi);
}

DEV f32x8 wmma_bf16(bf16x16 a, bf16x16 b, f32x8 c) {
  return __builtin_amdgcn_wmma_f32_16x16x32_bf16(
      /*neg_a=*/false, a, /*neg_b=*/false, b,
      /*c_mod=*/(short)0, c, /*reuse_a=*/false, /*reuse_b=*/false);
}

DEV f32x8 zero8() {
  f32x8 z = {0.f, 0.f, 0.f, 0.f, 0.f, 0.f, 0.f, 0.f};
  return z;
}

// ---------------------------------------------------------------------------
// f32 -> bf16 conversion, 8 elements per thread (b128 in, b128 out)
// ---------------------------------------------------------------------------
__global__ void k_cvt(const float* __restrict__ src, bf16* __restrict__ dst, int n8) {
  int i = blockIdx.x * blockDim.x + threadIdx.x;
  if (i >= n8) return;
  const u32x4* s = reinterpret_cast<const u32x4*>(src) + (size_t)i * 2;
  u32x4 x0 = s[0], x1 = s[1];
  u32x4 o;
  o[0] = (u32)bf16_bits(__builtin_bit_cast(float, x0[0])) |
         ((u32)bf16_bits(__builtin_bit_cast(float, x0[1])) << 16);
  o[1] = (u32)bf16_bits(__builtin_bit_cast(float, x0[2])) |
         ((u32)bf16_bits(__builtin_bit_cast(float, x0[3])) << 16);
  o[2] = (u32)bf16_bits(__builtin_bit_cast(float, x1[0])) |
         ((u32)bf16_bits(__builtin_bit_cast(float, x1[1])) << 16);
  o[3] = (u32)bf16_bits(__builtin_bit_cast(float, x1[2])) |
         ((u32)bf16_bits(__builtin_bit_cast(float, x1[3])) << 16);
  *(reinterpret_cast<u32x4*>(dst) + i) = o;
}

// ---------------------------------------------------------------------------
// Software-pipelined 16x64 GEMM tile: acc[0..3] += A(16xD) x B(Dx64)
// A row-major [16][D]; B source row-major [64][D] (i.e. multiplies by B^T).
// ---------------------------------------------------------------------------
DEV void gemm_tile_16x64(const bf16* __restrict__ arow,
                         const bf16* __restrict__ wrow,
                         f32x8& acc0, f32x8& acc1, f32x8& acc2, f32x8& acc3) {
  bf16x16 a_c = load_a(arow, Dn);
  bf16x16 b0 = load_b(wrow + 0 * 16 * Dn, Dn);
  bf16x16 b1 = load_b(wrow + 1 * 16 * Dn, Dn);
  bf16x16 b2 = load_b(wrow + 2 * 16 * Dn, Dn);
  bf16x16 b3 = load_b(wrow + 3 * 16 * Dn, Dn);
  for (int k0 = 0; k0 < Dn - 32; k0 += 32) {
    int kn = k0 + 32;
    __builtin_prefetch(arow + kn + 96, 0, 3);
    // issue next iteration's loads before consuming current fragments
    bf16x16 a_n = load_a(arow + kn, Dn);
    bf16x16 n0 = load_b(wrow + 0 * 16 * Dn + kn, Dn);
    bf16x16 n1 = load_b(wrow + 1 * 16 * Dn + kn, Dn);
    bf16x16 n2 = load_b(wrow + 2 * 16 * Dn + kn, Dn);
    bf16x16 n3 = load_b(wrow + 3 * 16 * Dn + kn, Dn);
    acc0 = wmma_bf16(a_c, b0, acc0);
    acc1 = wmma_bf16(a_c, b1, acc1);
    acc2 = wmma_bf16(a_c, b2, acc2);
    acc3 = wmma_bf16(a_c, b3, acc3);
    a_c = a_n; b0 = n0; b1 = n1; b2 = n2; b3 = n3;
  }
  acc0 = wmma_bf16(a_c, b0, acc0);
  acc1 = wmma_bf16(a_c, b1, acc1);
  acc2 = wmma_bf16(a_c, b2, acc2);
  acc3 = wmma_bf16(a_c, b3, acc3);
}

// ---------------------------------------------------------------------------
// Fused QKV projection: q/k/v = x @ W^T + b.
// Wave computes a 16-token x 64-col (one head) tile. 12288 waves.
// Q,K stored [B,H,S,64]; V stored transposed [B,H,64,S].
// ---------------------------------------------------------------------------
__global__ __launch_bounds__(256) void k_qkv(
    const bf16* __restrict__ xb,
    const bf16* __restrict__ Wqb, const bf16* __restrict__ Wkb, const bf16* __restrict__ Wvb,
    const float* __restrict__ bq, const float* __restrict__ bk, const float* __restrict__ bv,
    bf16* __restrict__ Qb, bf16* __restrict__ Kb, bf16* __restrict__ Vt) {
  int wave = blockIdx.x * 8 + (threadIdx.x >> 5);
  int mat  = wave >> 12;          // 0:Q 1:K 2:V (4096 waves each)
  int rem  = wave & 4095;
  int rt   = rem >> 4;            // row tile 0..255
  int h    = rem & 15;            // head
  const bf16*  W    = (mat == 0) ? Wqb : (mat == 1) ? Wkb : Wvb;
  const float* bias = (mat == 0) ? bq  : (mat == 1) ? bk  : bv;
  int row0 = rt * 16;             // token row in [0, B*S)

  f32x8 acc0 = zero8(), acc1 = zero8(), acc2 = zero8(), acc3 = zero8();
  gemm_tile_16x64(xb + (size_t)row0 * Dn, W + (size_t)(h * DK) * Dn,
                  acc0, acc1, acc2, acc3);

  int lane = threadIdx.x & 31;
  int colg = lane & 15, rsel = lane >> 4;
  f32x8 acc[4] = {acc0, acc1, acc2, acc3};
#pragma unroll
  for (int j = 0; j < 4; ++j) {
#pragma unroll
    for (int r = 0; r < 8; ++r) {
      int tok = row0 + r + 8 * rsel;
      int d   = j * 16 + colg;
      float val = acc[j][r] + bias[h * DK + d];
      bf16 hv = to_bf16(val);
      int b = tok >> 11, s = tok & (Sn - 1);
      size_t bh = (size_t)(b * Hn + h);
      if (mat == 0)      Qb[(bh * Sn + s) * DK + d] = hv;
      else if (mat == 1) Kb[(bh * Sn + s) * DK + d] = hv;
      else               Vt[(bh * DK + d) * Sn + s] = hv;
    }
  }
}

// ---------------------------------------------------------------------------
// Flash-attention for one 16-query tile of one (b,h). Pipelined:
//   preload K(0); per block: issue V loads -> score WMMAs -> softmax (V in
//   flight) -> issue next K -> P via LDS re-layout -> PV WMMAs.
// ---------------------------------------------------------------------------
DEV void attn_tile(const bf16* __restrict__ Kbh,   // K  [S][64]
                   const bf16* __restrict__ Vbh,   // V^T [64][S]
                   const bf16* __restrict__ Qp,    // Q tile base [16][64]
                   const int*  __restrict__ pmrow, // padding mask row (S ints)
                   bf16* __restrict__ ctx_row,     // ctx + (b*S+q0)*D + h*64
                   int q0, bf16* __restrict__ pb) {
  int lane = threadIdx.x & 31;
  int colg = lane & 15, rsel = lane >> 4;

  bf16x16 qa0 = load_a(Qp, DK);         // d = 0..31
  bf16x16 qa1 = load_a(Qp + 32, DK);    // d = 32..63

  f32x8 o0 = zero8(), o1 = zero8(), o2 = zero8(), o3 = zero8();
  float mrow[8], lrow[8];
#pragma unroll
  for (int r = 0; r < 8; ++r) { mrow[r] = -1.0e30f; lrow[r] = 0.f; }

  int nkb = (q0 + 47) >> 5;             // 32-key blocks covering keys <= q0+15

  // preload K fragments for block 0
  bf16x16 kf0 = load_b(Kbh, DK);
  bf16x16 kf1 = load_b(Kbh + 32, DK);
  bf16x16 kf2 = load_b(Kbh + 16 * DK, DK);
  bf16x16 kf3 = load_b(Kbh + 16 * DK + 32, DK);

  for (int kb = 0; kb < nkb; ++kb) {
    int ks = kb * 32;
    // issue V loads early; they complete after K (in-order) and are consumed
    // only after the LDS bounce, overlapping all the softmax VALU work.
    const bf16* Vp = Vbh + ks;
    bf16x16 vf0 = load_b(Vp + 0 * 16 * Sn, Sn);
    bf16x16 vf1 = load_b(Vp + 1 * 16 * Sn, Sn);
    bf16x16 vf2 = load_b(Vp + 2 * 16 * Sn, Sn);
    bf16x16 vf3 = load_b(Vp + 3 * 16 * Sn, Sn);

    f32x8 c0 = zero8(), c1 = zero8();
    c0 = wmma_bf16(qa0, kf0, c0);
    c0 = wmma_bf16(qa1, kf1, c0);
    c1 = wmma_bf16(qa0, kf2, c1);
    c1 = wmma_bf16(qa1, kf3, c1);

    int n0 = ks + colg, n1 = n0 + 16;
    bool p0m = (pmrow[n0] == 0);
    bool p1m = (pmrow[n1] == 0);

    float f0[8], f1[8], mx[8];
#pragma unroll
    for (int r = 0; r < 8; ++r) {
      int m = q0 + r + 8 * rsel;
      float v0 = c0[r] * SCALE; if (p0m || n0 > m) v0 = NEGV;
      float v1 = c1[r] * SCALE; if (p1m || n1 > m) v1 = NEGV;
      f0[r] = v0; f1[r] = v1; mx[r] = fmaxf(v0, v1);
    }
    // row max across the 16 lanes holding each row (C layout: row per half)
#pragma unroll
    for (int sh = 1; sh <= 8; sh <<= 1)
#pragma unroll
      for (int r = 0; r < 8; ++r)
        mx[r] = fmaxf(mx[r], __shfl_xor(mx[r], sh, 32));

    float al[8], ps[8];
#pragma unroll
    for (int r = 0; r < 8; ++r) {
      float mn = fmaxf(mrow[r], mx[r]);
      al[r] = __expf(mrow[r] - mn);
      mrow[r] = mn;
      f0[r] = __expf(f0[r] - mn);
      f1[r] = __expf(f1[r] - mn);
      ps[r] = f0[r] + f1[r];
    }
#pragma unroll
    for (int sh = 1; sh <= 8; sh <<= 1)
#pragma unroll
      for (int r = 0; r < 8; ++r)
        ps[r] += __shfl_xor(ps[r], sh, 32);
#pragma unroll
    for (int r = 0; r < 8; ++r) {
      lrow[r] = lrow[r] * al[r] + ps[r];
      o0[r] *= al[r]; o1[r] *= al[r]; o2[r] *= al[r]; o3[r] *= al[r];
    }

    // issue next block's K loads now; they fly during the LDS bounce + PV.
    if (kb + 1 < nkb) {
      const bf16* Kp = Kbh + (size_t)(ks + 32) * DK;
      kf0 = load_b(Kp, DK);
      kf1 = load_b(Kp + 32, DK);
      kf2 = load_b(Kp + 16 * DK, DK);
      kf3 = load_b(Kp + 16 * DK + 32, DK);
    }

    // P tile: C layout -> LDS -> A layout (per-wave region, no barrier needed)
#pragma unroll
    for (int r = 0; r < 8; ++r) {
      int m = r + 8 * rsel;
      pb[m * 16 + colg]       = to_bf16(f0[r]);   // tile0: keys ks..ks+15
      pb[256 + m * 16 + colg] = to_bf16(f1[r]);   // tile1: keys ks+16..ks+31
    }
    asm volatile("s_wait_dscnt 0" ::: "memory");
    bf16x16 pa = load_a_lds(pb);

    // O += P (16x32) x V (32x64)
    o0 = wmma_bf16(pa, vf0, o0);
    o1 = wmma_bf16(pa, vf1, o1);
    o2 = wmma_bf16(pa, vf2, o2);
    o3 = wmma_bf16(pa, vf3, o3);
  }

  // normalize and store context rows
#pragma unroll
  for (int r = 0; r < 8; ++r) {
    float inv = 1.0f / lrow[r];
    size_t base = (size_t)(r + 8 * rsel) * Dn;
    ctx_row[base + 0 * 16 + colg] = to_bf16(o0[r] * inv);
    ctx_row[base + 1 * 16 + colg] = to_bf16(o1[r] * inv);
    ctx_row[base + 2 * 16 + colg] = to_bf16(o2[r] * inv);
    ctx_row[base + 3 * 16 + colg] = to_bf16(o3[r] * inv);
  }
}

// ---------------------------------------------------------------------------
// Flash attention driver: one wave handles the causal-balanced pair of query
// tiles (qt, 127-qt) for one (b,h). 2048 waves, uniform work per wave.
// ---------------------------------------------------------------------------
__global__ __launch_bounds__(256) void k_attn(
    const bf16* __restrict__ Qb, const bf16* __restrict__ Kb,
    const bf16* __restrict__ Vt, const int* __restrict__ pmask,
    bf16* __restrict__ ctx) {
  __shared__ bf16 Pl[8 * 512];          // per-wave 2x(16x16) bf16 P tiles
  int wv   = threadIdx.x >> 5;
  int wave = blockIdx.x * 8 + wv;
  int bh   = wave >> 6;                 // 0..31
  int qp   = wave & 63;                 // tile pair index
  int b    = bh >> 4;
  int h    = bh & 15;
  bf16* pb = Pl + wv * 512;

  const bf16* Kbh = Kb + (size_t)bh * Sn * DK;
  const bf16* Vbh = Vt + (size_t)bh * DK * Sn;
  const int*  pmr = pmask + b * Sn;

  int qts[2] = {qp, 127 - qp};
#pragma unroll
  for (int t = 0; t < 2; ++t) {
    int q0 = qts[t] * 16;
    const bf16* Qp = Qb + ((size_t)bh * Sn + q0) * DK;
    bf16* crow = ctx + ((size_t)b * Sn + q0) * Dn + h * DK;
    attn_tile(Kbh, Vbh, Qp, pmr, crow, q0, pb);
  }
}

// ---------------------------------------------------------------------------
// Output projection: out = ctx @ Wo^T + bo (f32 out). 4096 waves.
// ---------------------------------------------------------------------------
__global__ __launch_bounds__(256) void k_oproj(
    const bf16* __restrict__ ctx, const bf16* __restrict__ Wob,
    const float* __restrict__ bo, float* __restrict__ out) {
  int wave = blockIdx.x * 8 + (threadIdx.x >> 5);
  int rt = wave >> 4;            // row tile 0..255
  int ct = wave & 15;            // 64-col tile
  int row0 = rt * 16;

  f32x8 acc0 = zero8(), acc1 = zero8(), acc2 = zero8(), acc3 = zero8();
  gemm_tile_16x64(ctx + (size_t)row0 * Dn, Wob + (size_t)(ct * 64) * Dn,
                  acc0, acc1, acc2, acc3);

  int lane = threadIdx.x & 31;
  int colg = lane & 15, rsel = lane >> 4;
  f32x8 acc[4] = {acc0, acc1, acc2, acc3};
#pragma unroll
  for (int j = 0; j < 4; ++j) {
#pragma unroll
    for (int r = 0; r < 8; ++r) {
      int m = row0 + r + 8 * rsel;
      int n = ct * 64 + j * 16 + colg;
      out[(size_t)m * Dn + n] = acc[j][r] + bo[n];
    }
  }
}

// ---------------------------------------------------------------------------
// Launch
// ---------------------------------------------------------------------------
extern "C" void kernel_launch(void* const* d_in, const int* in_sizes, int n_in,
                              void* d_out, int out_size, void* d_ws, size_t ws_size,
                              hipStream_t stream) {
  (void)in_sizes; (void)n_in; (void)out_size; (void)ws_size;
  const float* x  = (const float*)d_in[0];
  const int*   pm = (const int*)  d_in[1];
  const float* Wq = (const float*)d_in[2];
  const float* bq = (const float*)d_in[3];
  const float* Wk = (const float*)d_in[4];
  const float* bk = (const float*)d_in[5];
  const float* Wv = (const float*)d_in[6];
  const float* bv = (const float*)d_in[7];
  const float* Wo = (const float*)d_in[8];
  const float* bo = (const float*)d_in[9];
  float* out = (float*)d_out;

  const size_t NX = (size_t)Bn * Sn * Dn;   // 4,194,304
  const size_t NW = (size_t)Dn * Dn;        // 1,048,576

  bf16* w   = (bf16*)d_ws;
  bf16* xb  = w; w += NX;
  bf16* Wqb = w; w += NW;
  bf16* Wkb = w; w += NW;
  bf16* Wvb = w; w += NW;
  bf16* Wob = w; w += NW;
  bf16* Qb  = w; w += NX;
  bf16* Kb  = w; w += NX;
  bf16* Vt  = w; w += NX;
  bf16* ctx = w; w += NX;      // total ~48 MB of d_ws

  int nx8 = (int)(NX / 8), nw8 = (int)(NW / 8);
  k_cvt<<<(nx8 + 255) / 256, 256, 0, stream>>>(x,  xb,  nx8);
  k_cvt<<<(nw8 + 255) / 256, 256, 0, stream>>>(Wq, Wqb, nw8);
  k_cvt<<<(nw8 + 255) / 256, 256, 0, stream>>>(Wk, Wkb, nw8);
  k_cvt<<<(nw8 + 255) / 256, 256, 0, stream>>>(Wv, Wvb, nw8);
  k_cvt<<<(nw8 + 255) / 256, 256, 0, stream>>>(Wo, Wob, nw8);

  k_qkv  <<<1536, 256, 0, stream>>>(xb, Wqb, Wkb, Wvb, bq, bk, bv, Qb, Kb, Vt);
  k_attn <<< 256, 256, 0, stream>>>(Qb, Kb, Vt, pm, ctx);
  k_oproj<<< 512, 256, 0, stream>>>(ctx, Wob, bo, out);
}
